// TwoBranchGNN_31490700214324
// MI455X (gfx1250) — matrix operand order repaired and checked by reading, not compiled, not run on hardware
//
#include <hip/hip_runtime.h>

// ---------------- problem constants (match reference) ----------------
#define NODES   50000
#define HID     128
#define NCLS    40

// WMMA vector types (gfx1250, wave32)
typedef __attribute__((ext_vector_type(16))) __bf16 v16bf;
typedef __attribute__((ext_vector_type(8)))  float  v8f;

union V16BF {
  v16bf v;
  unsigned short u[16];
  uint4 q[2];
};

union U8BF {
  uint4 q;
  unsigned short u[8];
};

// f32 -> bf16 (round-to-nearest-even), bf16 -> f32
__device__ __forceinline__ unsigned short f2bf(float f) {
  unsigned int u = __float_as_uint(f);
  unsigned int r = u + 0x7FFFu + ((u >> 16) & 1u);
  return (unsigned short)(r >> 16);
}
__device__ __forceinline__ float bf2f(unsigned short h) {
  return __uint_as_float((unsigned int)h << 16);
}

// B operand: pre-swizzled packed bf16, 32 contiguous bytes per lane
__device__ __forceinline__ v16bf load_b_packed(const unsigned short* __restrict__ p) {
  V16BF b;
  b.q[0] = *(const uint4*)(p + 0);
  b.q[1] = *(const uint4*)(p + 8);
  return b.v;
}

// A operand from a row-major bf16 16x128 tile (base = tile + row*HID):
// lane L: row m = L&15, half = L>>4; element j -> K = (j>>3)*16 + half*8 + (j&7)
// => two contiguous 16B runs at columns [kc*32+half*8] and [kc*32+16+half*8]
__device__ __forceinline__ v16bf load_a_tile(const unsigned short* p) {
  V16BF a;
  a.q[0] = *(const uint4*)(p);       // K run 0..7
  a.q[1] = *(const uint4*)(p + 16);  // K run 16..23 (+16 columns)
  return a.v;
}

// ---------------- f32 -> bf16 bulk convert (for x0) ----------------
__global__ __launch_bounds__(256)
void cvt_bf16_kernel(const float* __restrict__ src, unsigned short* __restrict__ dst, int ntot8) {
  const int t = blockIdx.x * blockDim.x + threadIdx.x;
  if (t >= ntot8) return;
  float4 f0 = *((const float4*)src + 2 * t + 0);
  float4 f1 = *((const float4*)src + 2 * t + 1);
  U8BF o;
  o.u[0] = f2bf(f0.x); o.u[1] = f2bf(f0.y); o.u[2] = f2bf(f0.z); o.u[3] = f2bf(f0.w);
  o.u[4] = f2bf(f1.x); o.u[5] = f2bf(f1.y); o.u[6] = f2bf(f1.z); o.u[7] = f2bf(f1.w);
  *((uint4*)dst + t) = o.q;
}

// ---------------- weight pre-swizzle ----------------
// B operand layout (32x16 bf16 tile): lane L: col n = L&15, half = L>>4;
// element j -> K = (j>>3)*16 + half*8 + (j&7).   Packed index: (((ct*4+kc)*32 + lane)*16 + j)
__global__ __launch_bounds__(512)
void pack_w_kernel(const float* __restrict__ W, unsigned short* __restrict__ P, int ncols) {
  const int ct   = blockIdx.x >> 2;
  const int kc   = blockIdx.x & 3;
  const int lane = threadIdx.x >> 4;
  const int j    = threadIdx.x & 15;
  const int half = lane >> 4;
  const int n    = ct * 16 + (lane & 15);
  const int k    = kc * 32 + (j >> 3) * 16 + half * 8 + (j & 7);
  float w = 0.f;
  if (n < ncols) w = W[(size_t)k * ncols + n];
  P[(((size_t)(ct * 4 + kc)) * 32 + (size_t)lane) * 16 + j] = f2bf(w);
}

// ---------------- SpMM: aggr[n] = sum_e val[e] * x[(remap?)col[e]]  (bf16 in/out) ----
__device__ __forceinline__ int lbound(const int* __restrict__ a, int n, int key) {
  int lo = 0, hi = n;
  while (lo < hi) { int mid = (lo + hi) >> 1; if (a[mid] < key) lo = mid + 1; else hi = mid; }
  return lo;
}

// one wave per node; rows are sorted so each node's edges are a contiguous span
__global__ __launch_bounds__(256)
void spmm_kernel(const unsigned short* __restrict__ x, const int* __restrict__ rows,
                 const int* __restrict__ cols, const float* __restrict__ vals,
                 const int* __restrict__ remap, unsigned short* __restrict__ aggr, int nE) {
  const int node = (int)(((size_t)blockIdx.x * blockDim.x + threadIdx.x) >> 5);
  const int lane = threadIdx.x & 31;
  if (node >= NODES) return;
  const int lo = lbound(rows, nE, node);
  const int hi = lbound(rows, nE, node + 1);
  float a0 = 0.f, a1 = 0.f, a2 = 0.f, a3 = 0.f;
  for (int e = lo; e < hi; ++e) {
    int c = cols[e];
    if (remap) c = remap[c];
    const float v = vals[e];
    if (e + 1 < hi) {  // prefetch next edge's feature row (global_prefetch_b8)
      int cn = cols[e + 1];
      if (remap) cn = remap[cn];
      __builtin_prefetch(x + (size_t)cn * HID + lane * 4, 0, 0);
    }
    const uint2 w = *((const uint2*)(x + (size_t)c * HID) + lane);  // 4 bf16
    a0 += v * bf2f((unsigned short)(w.x & 0xffff));
    a1 += v * bf2f((unsigned short)(w.x >> 16));
    a2 += v * bf2f((unsigned short)(w.y & 0xffff));
    a3 += v * bf2f((unsigned short)(w.y >> 16));
  }
  uint2 o;
  o.x = (unsigned)f2bf(a0) | ((unsigned)f2bf(a1) << 16);
  o.y = (unsigned)f2bf(a2) | ((unsigned)f2bf(a3) << 16);
  *((uint2*)(aggr + (size_t)node * HID) + lane) = o;
}

// ---------------- x_mix0 = x0*r + x0[idx]*(1-r)  (bf16) ----------------
__global__ __launch_bounds__(256)
void gather_mix_kernel(const unsigned short* __restrict__ x0, const int* __restrict__ idx,
                       const float* __restrict__ mixr, unsigned short* __restrict__ outp) {
  const size_t t = (size_t)blockIdx.x * blockDim.x + threadIdx.x;
  const int node  = (int)(t >> 5);
  const int chunk = (int)(t & 31);
  if (node >= NODES) return;
  const float r = mixr[0];
  const int   s = idx[node];
  const uint2 a = *((const uint2*)(x0 + (size_t)node * HID) + chunk);
  const uint2 b = *((const uint2*)(x0 + (size_t)s    * HID) + chunk);
  float m0 = bf2f((unsigned short)(a.x & 0xffff)) * r + bf2f((unsigned short)(b.x & 0xffff)) * (1.f - r);
  float m1 = bf2f((unsigned short)(a.x >> 16))    * r + bf2f((unsigned short)(b.x >> 16))    * (1.f - r);
  float m2 = bf2f((unsigned short)(a.y & 0xffff)) * r + bf2f((unsigned short)(b.y & 0xffff)) * (1.f - r);
  float m3 = bf2f((unsigned short)(a.y >> 16))    * r + bf2f((unsigned short)(b.y >> 16))    * (1.f - r);
  uint2 o;
  o.x = (unsigned)f2bf(m0) | ((unsigned)f2bf(m1) << 16);
  o.y = (unsigned)f2bf(m2) | ((unsigned)f2bf(m3) << 16);
  *((uint2*)(outp + (size_t)node * HID) + chunk) = o;
}

// ---------------- fused SAGE GEMM: out = epi(aggr@Wl + xdst@Wr + b) ----------------
// grid = NODES/16 blocks, 256 threads = 8 waves; wave w owns col-tile ct = w.
// A1/A2 16x128 bf16 tiles are async-staged into LDS once per block (shared by all 8 waves).
// mode 0: out = relu(acc + b);  mode 1: out = prev*r + relu(acc + b)*(1-r)
__global__ __launch_bounds__(256)
void sage_gemm_kernel(const unsigned short* __restrict__ A1, const unsigned short* __restrict__ B1,
                      const unsigned short* __restrict__ A2, const unsigned short* __restrict__ B2,
                      const float* __restrict__ bias, unsigned short* __restrict__ outp,
                      const unsigned short* __restrict__ prev, const float* __restrict__ mixr,
                      int mode) {
  __shared__ __align__(16) unsigned short sA[2 * 16 * HID];  // 8KB: [A1 tile | A2 tile]
  const int tid = threadIdx.x;
  const size_t nodeBase = (size_t)blockIdx.x * 16;

  // ---- async stage: 256 threads x one 16B segment of each tile ----
  {
    const int row = tid >> 4;                       // 0..15
    const int sc  = tid & 15;                       // 16B segment within 256B row
    const unsigned ldsBase = (unsigned)(unsigned long long)(&sA[0]);
    const unsigned lds0 = ldsBase + (unsigned)(tid * 16);
    const unsigned lds1 = lds0 + (unsigned)(16 * HID * 2);   // +4096B
    const unsigned long long g1 =
        (unsigned long long)(const void*)(A1 + (nodeBase + row) * HID + sc * 8);
    const unsigned long long g2 =
        (unsigned long long)(const void*)(A2 + (nodeBase + row) * HID + sc * 8);
    asm volatile("global_load_async_to_lds_b128 %0, %1, off" :: "v"(lds0), "v"(g1) : "memory");
    asm volatile("global_load_async_to_lds_b128 %0, %1, off" :: "v"(lds1), "v"(g2) : "memory");
    asm volatile("s_wait_asynccnt 0" ::: "memory");
  }
  __syncthreads();

  const int ct   = tid >> 5;
  const int lane = tid & 31;
  const int half = lane >> 4;
  const int ml   = lane & 15;

  const unsigned short* sA1 = sA;
  const unsigned short* sA2 = sA + 16 * HID;
  const unsigned short* b1  = B1 + ((size_t)(ct * 4) * 32 + (size_t)lane) * 16;
  const unsigned short* b2  = B2 + ((size_t)(ct * 4) * 32 + (size_t)lane) * 16;
  const int aoff = ml * HID + half * 8;

  v8f acc = {0.f, 0.f, 0.f, 0.f, 0.f, 0.f, 0.f, 0.f};
#pragma unroll
  for (int kc = 0; kc < 4; ++kc) {
    v16bf av = load_a_tile(sA1 + aoff + kc * 32);          // ds_load_b128 x2
    v16bf bv = load_b_packed(b1 + (size_t)kc * 512);
    acc = __builtin_amdgcn_wmma_f32_16x16x32_bf16(false, av, false, bv, (short)0, acc, false, false);
  }
#pragma unroll
  for (int kc = 0; kc < 4; ++kc) {
    v16bf av = load_a_tile(sA2 + aoff + kc * 32);
    v16bf bv = load_b_packed(b2 + (size_t)kc * 512);
    acc = __builtin_amdgcn_wmma_f32_16x16x32_bf16(false, av, false, bv, (short)0, acc, false, false);
  }

  const int   nCol = ct * 16 + ml;        // C/D layout: lane holds column lane&15
  const float bb   = bias[nCol];
  float r = 0.f;
  if (mode == 1) r = mixr[0];
#pragma unroll
  for (int rr = 0; rr < 8; ++rr) {
    const int mrow = half * 8 + rr;       // C/D layout: VGPR rr -> row half*8+rr
    float vv = fmaxf(acc[rr] + bb, 0.f);
    const size_t o = (nodeBase + mrow) * HID + nCol;
    if (mode == 1) vv = bf2f(prev[o]) * r + vv * (1.f - r);
    outp[o] = f2bf(vv);
  }
}

// ---------------- classifier GEMM: logits = xmix @ W_out + b_out ----------------
// 3 col-tiles (40 classes padded to 48), 3 waves/block; A read directly (bf16)
__global__ __launch_bounds__(96)
void out_gemm_kernel(const unsigned short* __restrict__ A, const unsigned short* __restrict__ B,
                     const float* __restrict__ bias, float* __restrict__ logits) {
  const int ct   = threadIdx.x >> 5;
  const int lane = threadIdx.x & 31;
  const int half = lane >> 4;
  const int ml   = lane & 15;
  const size_t nodeBase = (size_t)blockIdx.x * 16;

  const unsigned short* a = A + (nodeBase + ml) * HID + half * 8;
  const unsigned short* b = B + ((size_t)(ct * 4) * 32 + (size_t)lane) * 16;

  v8f acc = {0.f, 0.f, 0.f, 0.f, 0.f, 0.f, 0.f, 0.f};
#pragma unroll
  for (int kc = 0; kc < 4; ++kc) {
    v16bf av = load_a_tile(a + kc * 32);
    v16bf bv = load_b_packed(b + (size_t)kc * 512);
    acc = __builtin_amdgcn_wmma_f32_16x16x32_bf16(false, av, false, bv, (short)0, acc, false, false);
  }

  const int nCol = ct * 16 + ml;
  if (nCol < NCLS) {
    const float bb = bias[nCol];
#pragma unroll
    for (int rr = 0; rr < 8; ++rr) {
      const int mrow = half * 8 + rr;
      logits[(nodeBase + mrow) * NCLS + nCol] = acc[rr] + bb;
    }
  }
}

// ---------------- log-softmax over 40 classes, one thread per node ----------------
__global__ __launch_bounds__(256)
void logsoftmax_kernel(const float* __restrict__ logits, float* __restrict__ outp) {
  const int n = blockIdx.x * blockDim.x + threadIdx.x;
  if (n >= NODES) return;
  const float* p = logits + (size_t)n * NCLS;
  float mx = p[0];
#pragma unroll 4
  for (int c = 1; c < NCLS; ++c) mx = fmaxf(mx, p[c]);
  float s = 0.f;
#pragma unroll 4
  for (int c = 0; c < NCLS; ++c) s += __expf(p[c] - mx);
  const float ls = __logf(s);
#pragma unroll 4
  for (int c = 0; c < NCLS; ++c) outp[(size_t)n * NCLS + c] = p[c] - mx - ls;
}

// ---------------- launch ----------------
extern "C" void kernel_launch(void* const* d_in, const int* in_sizes, int n_in,
                              void* d_out, int out_size, void* d_ws, size_t ws_size,
                              hipStream_t stream) {
  const float* x0f   = (const float*)d_in[0];
  const int*   aRow  = (const int*)  d_in[1];
  const int*   aCol  = (const int*)  d_in[2];
  const float* aVal  = (const float*)d_in[3];
  const int*   bRow  = (const int*)  d_in[4];
  const int*   bCol  = (const int*)  d_in[5];
  const float* bVal  = (const float*)d_in[6];
  const float* mixr  = (const float*)d_in[7];
  const int*   remap = (const int*)  d_in[8];
  const float* Wl    = (const float*)d_in[9];
  const float* Wr    = (const float*)d_in[10];
  const float* bs    = (const float*)d_in[11];
  const float* Wo    = (const float*)d_in[12];
  const float* bo    = (const float*)d_in[13];
  float* out = (float*)d_out;
  const int EA = in_sizes[1];
  const int EB = in_sizes[4];
  (void)n_in; (void)out_size; (void)ws_size;

  // carve workspace
  char* ws = (char*)d_ws;
  size_t off = 0;
  auto carve = [&](size_t bytes) -> void* {
    void* p = ws + off;
    off += (bytes + 255) & ~(size_t)255;
    return p;
  };
  const size_t featH = (size_t)NODES * HID * sizeof(unsigned short);  // bf16 features
  unsigned short* x0b   = (unsigned short*)carve(featH);
  unsigned short* aggr  = (unsigned short*)carve(featH);
  unsigned short* x1    = (unsigned short*)carve(featH);
  unsigned short* x2    = (unsigned short*)carve(featH);
  unsigned short* xnew  = (unsigned short*)carve(featH);
  unsigned short* mixA  = (unsigned short*)carve(featH);
  unsigned short* mixB  = (unsigned short*)carve(featH);
  float* logits = (float*)carve((size_t)NODES * NCLS * sizeof(float));
  unsigned short* pk = (unsigned short*)carve((size_t)(6 * 16384 + 6144) * sizeof(unsigned short));

  unsigned short* pWl[3]; unsigned short* pWr[3];
  for (int l = 0; l < 3; ++l) { pWl[l] = pk + (size_t)l * 16384; pWr[l] = pk + (size_t)(3 + l) * 16384; }
  unsigned short* pWo = pk + (size_t)6 * 16384;

  // x0 -> bf16 once; pre-swizzle weights to packed bf16 B-operand layout
  cvt_bf16_kernel<<<dim3(NODES * HID / 8 / 256), dim3(256), 0, stream>>>(x0f, x0b, NODES * HID / 8);
  for (int l = 0; l < 3; ++l) {
    pack_w_kernel<<<dim3(32), dim3(512), 0, stream>>>(Wl + (size_t)l * HID * HID, pWl[l], HID);
    pack_w_kernel<<<dim3(32), dim3(512), 0, stream>>>(Wr + (size_t)l * HID * HID, pWr[l], HID);
  }
  pack_w_kernel<<<dim3(12), dim3(512), 0, stream>>>(Wo, pWo, NCLS);

  const dim3 gSpmm(NODES / 8), bSpmm(256);   // one wave per node
  const dim3 gGemm(NODES / 16), bGemm(256);  // one block per 16-row tile, 8 waves = 8 col tiles

  // branch A pre-aggregation stack: aggr_xs = [x0b, x1, x2]
  spmm_kernel<<<gSpmm, bSpmm, 0, stream>>>(x0b, aRow, aCol, aVal, nullptr, aggr, EA);
  sage_gemm_kernel<<<gGemm, bGemm, 0, stream>>>(aggr, pWl[0], x0b, pWr[0], bs + 0 * HID, x1, nullptr, mixr, 0);
  spmm_kernel<<<gSpmm, bSpmm, 0, stream>>>(x1, aRow, aCol, aVal, nullptr, aggr, EA);
  sage_gemm_kernel<<<gGemm, bGemm, 0, stream>>>(aggr, pWl[1], x1, pWr[1], bs + 1 * HID, x2, nullptr, mixr, 0);

  // initial mix: x_mix = x0*r + x0[idx]*(1-r)
  gather_mix_kernel<<<dim3(NODES / 8), dim3(256), 0, stream>>>(x0b, remap, mixr, mixA);

  // mix loop: branch-B gather fused into SpMM via remap; blend fused into epilogue
  const unsigned short* xs[3] = {x0b, x1, x2};
  unsigned short* mixCur = mixA;
  unsigned short* mixNxt = mixB;
  for (int i = 0; i < 3; ++i) {
    spmm_kernel<<<gSpmm, bSpmm, 0, stream>>>(xs[i], aRow, aCol, aVal, nullptr, aggr, EA);
    sage_gemm_kernel<<<gGemm, bGemm, 0, stream>>>(aggr, pWl[i], mixCur, pWr[i], bs + (size_t)i * HID,
                                                  xnew, nullptr, mixr, 0);
    spmm_kernel<<<gSpmm, bSpmm, 0, stream>>>(xs[i], bRow, bCol, bVal, remap, aggr, EB);
    sage_gemm_kernel<<<gGemm, bGemm, 0, stream>>>(aggr, pWl[i], mixCur, pWr[i], bs + (size_t)i * HID,
                                                  mixNxt, xnew, mixr, 1);
    unsigned short* t = mixCur; mixCur = mixNxt; mixNxt = t;
  }

  // classifier + log-softmax
  out_gemm_kernel<<<gGemm, dim3(96), 0, stream>>>(mixCur, pWo, bo, logits);
  logsoftmax_kernel<<<dim3((NODES + 255) / 256), dim3(256), 0, stream>>>(logits, out);
}
